// Live_Gandalf_Model_V9_57466662420624
// MI455X (gfx1250) — matrix operand order repaired and checked by reading, not compiled, not run on hardware
//
#include <hip/hip_runtime.h>

#define Bsz 4096
#define Fn 256
#define L0c 64
#define BN 8
#define BD 128
#define HID 256
#define NE 4
#define TOPK 2
#define PROJ_IN 33664   // 128 + 768 + 256*128
#define KB 32
#define NKB (PROJ_IN / KB)   // 1052

typedef __attribute__((ext_vector_type(16))) __bf16 v16bf;
typedef __attribute__((ext_vector_type(8)))  float  v8f;
typedef unsigned int v4u __attribute__((ext_vector_type(4)));
typedef int v4i __attribute__((ext_vector_type(4)));
typedef int v8i __attribute__((ext_vector_type(8)));

union BF16x16 { v16bf v; unsigned int u[8]; uint4 q[2]; };
union Pack8 { uint4 q; unsigned short s[8]; };

__device__ inline unsigned short f2bf(float x) {
  unsigned int u = __float_as_uint(x);
  unsigned int r = u + 0x7FFFu + ((u >> 16) & 1u);   // round-to-nearest-even
  return (unsigned short)(r >> 16);
}

// ---------------------------------------------------------------------------
// Kernel A: sanitize + auto_dis (per-feature MLP -> softmax -> embedding mix)
// writes the full concatenated proj input row [asr | mm | y] as bf16.
// One block per batch row, one thread per feature.
// ---------------------------------------------------------------------------
__global__ __launch_bounds__(256) void k_autodis(
    const float* __restrict__ stat, const float* __restrict__ asr,
    const float* __restrict__ mm,
    const float* __restrict__ l1w, const float* __restrict__ l1b,
    const float* __restrict__ l2w, const float* __restrict__ l2b,
    const float* __restrict__ tau, const float* __restrict__ emb,
    unsigned short* __restrict__ xcb) {
  const int b = blockIdx.x;
  const int f = threadIdx.x;
  const size_t row = (size_t)b * PROJ_IN;
  for (int i = f; i < 896; i += 256) {
    float v = (i < 128) ? asr[b * 128 + i] : mm[b * 768 + (i - 128)];
    xcb[row + i] = f2bf(v);
  }
  float x = stat[b * Fn + f];
  if (!(x == x)) x = 0.0f;        // nan -> 0
  if (x < 0.0f) x = 0.0f;         // negatives -> 0
  float h2[BN];
#pragma unroll
  for (int k = 0; k < BN; ++k) h2[k] = l2b[f * BN + k];
  const float* w1 = l1w + (size_t)f * 3 * L0c;
  const float* b1 = l1b + (size_t)f * L0c;
  const float* w2 = l2w + (size_t)f * L0c * BN;
  for (int l = 0; l < L0c; ++l) {
    float ws = w1[l] + w1[L0c + l] + w1[2 * L0c + l];
    float hl = fmaf(x, ws, b1[l]);
    hl = hl > 0.0f ? hl : 0.01f * hl;                  // leaky_relu
#pragma unroll
    for (int k = 0; k < BN; ++k) h2[k] = fmaf(hl, w2[l * BN + k], h2[k]);
  }
  float z[BN];
  float zmax = -3.4e38f;
#pragma unroll
  for (int k = 0; k < BN; ++k) {
    float v = h2[k] > 0.0f ? h2[k] : 0.01f * h2[k];
    z[k] = v * tau[f * BN + k];
    zmax = fmaxf(zmax, z[k]);
  }
  float zs = 0.0f;
#pragma unroll
  for (int k = 0; k < BN; ++k) { z[k] = __expf(z[k] - zmax); zs += z[k]; }
  float inv = 1.0f / zs;
#pragma unroll
  for (int k = 0; k < BN; ++k) z[k] *= inv;
  const float* ef = emb + (size_t)f * BN * BD;
  unsigned short* yout = xcb + row + 896 + (size_t)f * BD;
  for (int d0 = 0; d0 < BD; d0 += 8) {
    Pack8 pk;
#pragma unroll
    for (int d = 0; d < 8; ++d) {
      float acc = 0.0f;
#pragma unroll
      for (int k = 0; k < BN; ++k) acc = fmaf(z[k], ef[k * BD + d0 + d], acc);
      pk.s[d] = f2bf(acc);
    }
    *(uint4*)(yout + d0) = pk.q;   // 16B-aligned
  }
}

// ---------------------------------------------------------------------------
// Pre-swizzle proj_w [33664,256] into exact WMMA B-fragment order, bf16.
// dword d -> (kbi, jt, lane, v); dword = pack(B[k0][n], B[k0+1][n]),
// k0 = kbi*32 + 8*kh + 2*(v&3) + 16*(v>>2), n = jt*16 + ln.
// ---------------------------------------------------------------------------
__global__ void k_swz_proj_w(const float* __restrict__ pw,
                             unsigned int* __restrict__ wsw) {
  int d = blockIdx.x * blockDim.x + threadIdx.x;
  if (d >= NKB * 4096) return;
  int v = d & 7, lane = (d >> 3) & 31, jt = (d >> 8) & 15, kbi = d >> 12;
  int k = kbi * KB + 8 * (lane >> 4) + 2 * (v & 3) + 16 * (v >> 2);
  int n = jt * 16 + (lane & 15);
  unsigned int lo = f2bf(pw[(size_t)k * HID + n]);
  unsigned int hi = f2bf(pw[(size_t)(k + 1) * HID + n]);
  wsw[d] = lo | (hi << 16);
}

// e_w1 [NE,256,128] -> B-fragment order: d -> (e,jt,ks,lane,v)
__global__ void k_swz_ew1(const float* __restrict__ ew1,
                          unsigned int* __restrict__ esw) {
  int d = blockIdx.x * blockDim.x + threadIdx.x;
  if (d >= NE * 8 * 8 * 32 * 8) return;
  int v = d & 7, lane = (d >> 3) & 31, ks = (d >> 8) & 7, jt = (d >> 11) & 7,
      e = d >> 14;
  int k = ks * KB + 8 * (lane >> 4) + 2 * (v & 3) + 16 * (v >> 2);
  int h = jt * 16 + (lane & 15);
  unsigned int lo = f2bf(ew1[((size_t)(e * HID) + k) * 128 + h]);
  unsigned int hi = f2bf(ew1[((size_t)(e * HID) + k + 1) * 128 + h]);
  esw[d] = lo | (hi << 16);
}

// ---------------------------------------------------------------------------
// Kernel B: proj GEMM  xp[4096,256] = relu(xc[4096,33664] @ W + b)
// bf16 WMMA 16x16x32.  No LDS, no barriers; A fragments straight from the
// row-major bf16 activations, B fragments from the pre-swizzled weights.
// Also emits a bf16 mirror of xp to feed the TDM tile load in k_moe.
// ---------------------------------------------------------------------------
__global__ __launch_bounds__(256) void k_proj(
    const unsigned short* __restrict__ xcb, const unsigned int* __restrict__ wsw,
    const float* __restrict__ pb, float* __restrict__ xp,
    unsigned short* __restrict__ xph) {
  const int tid = threadIdx.x;
  const int m0 = blockIdx.x * 128;
  const int nt0 = blockIdx.y * 4;
  const int lane = tid & 31, w = tid >> 5;
  const int wm = w & 3, wn = w >> 2;          // 4x2 waves over 128x64
  const int kh = lane >> 4, ln = lane & 15;
  v8f c[2][2] = {};
  const unsigned short* a0 =
      xcb + (size_t)(m0 + wm * 32 + ln) * PROJ_IN + 8 * kh;
  const unsigned short* a1 = a0 + (size_t)16 * PROJ_IN;
  const unsigned int* b0 = wsw + ((size_t)(nt0 + wn * 2) * 32 + lane) * 8;
  const unsigned int* b1 = wsw + ((size_t)(nt0 + wn * 2 + 1) * 32 + lane) * 8;
  const size_t bstep = 4096;

#pragma unroll 2
  for (int kb = 0; kb < PROJ_IN; kb += KB) {
    BF16x16 a[2], bf[2];
    a[0].q[0] = *(const uint4*)(a0 + kb);
    a[0].q[1] = *(const uint4*)(a0 + kb + 16);
    a[1].q[0] = *(const uint4*)(a1 + kb);
    a[1].q[1] = *(const uint4*)(a1 + kb + 16);
    bf[0].q[0] = *(const uint4*)(b0);
    bf[0].q[1] = *(const uint4*)(b0 + 4);
    bf[1].q[0] = *(const uint4*)(b1);
    bf[1].q[1] = *(const uint4*)(b1 + 4);
    __builtin_prefetch(a0 + kb + 8 * KB, 0, 3);        // near-scope prefetch
    __builtin_prefetch((const void*)(b0 + 8 * bstep), 0, 3);
    b0 += bstep; b1 += bstep;
#pragma unroll
    for (int i = 0; i < 2; ++i)
#pragma unroll
      for (int j = 0; j < 2; ++j)
        c[i][j] = __builtin_amdgcn_wmma_f32_16x16x32_bf16(
            false, a[i].v, false, bf[j].v, (short)0, c[i][j], false, false);
  }
#pragma unroll
  for (int i = 0; i < 2; ++i)
#pragma unroll
    for (int j = 0; j < 2; ++j)
#pragma unroll
      for (int r = 0; r < 8; ++r) {
        int m = m0 + wm * 32 + i * 16 + r + 8 * kh;
        int n = (nt0 + wn * 2 + j) * 16 + ln;
        float v = c[i][j][r] + pb[n];
        v = v > 0.0f ? v : 0.0f;
        xp[(size_t)m * HID + n] = v;
        xph[(size_t)m * HID + n] = f2bf(v);
      }
}

// ---------------------------------------------------------------------------
// gate GEMV + batch-axis softmax statistics
// ---------------------------------------------------------------------------
__global__ void k_gate(const float* __restrict__ xp, const float* __restrict__ wg,
                       float* __restrict__ g) {
  int n = blockIdx.x * blockDim.x + threadIdx.x;
  if (n >= Bsz) return;
  float acc[NE] = {};
  const float* xr = xp + (size_t)n * HID;
  for (int k = 0; k < HID; ++k) {
    float x = xr[k];
#pragma unroll
    for (int e = 0; e < NE; ++e) acc[e] = fmaf(x, wg[k * NE + e], acc[e]);
  }
#pragma unroll
  for (int e = 0; e < NE; ++e) g[n * NE + e] = acc[e];
}

__global__ __launch_bounds__(256) void k_gate_stats(const float* __restrict__ g,
                                                    float* __restrict__ cm,
                                                    float* __restrict__ cs) {
  __shared__ float red[256];
  int tid = threadIdx.x;
  for (int e = 0; e < NE; ++e) {
    float m = -3.4e38f;
    for (int n = tid; n < Bsz; n += 256) m = fmaxf(m, g[n * NE + e]);
    red[tid] = m; __syncthreads();
    for (int s = 128; s > 0; s >>= 1) {
      if (tid < s) red[tid] = fmaxf(red[tid], red[tid + s]);
      __syncthreads();
    }
    m = red[0]; __syncthreads();
    float s = 0.0f;
    for (int n = tid; n < Bsz; n += 256) s += __expf(g[n * NE + e] - m);
    red[tid] = s; __syncthreads();
    for (int q = 128; q > 0; q >>= 1) {
      if (tid < q) red[tid] += red[tid + q];
      __syncthreads();
    }
    if (tid == 0) { cm[e] = m; cs[e] = red[0]; }
    __syncthreads();
  }
}

// ---------------------------------------------------------------------------
// Kernel D: expert MLPs (WMMA, one wave per expert, 16 tokens per block)
// + top-2 gather/combine.  The 16x256 bf16 activation tile is staged into LDS
// via the Tensor Data Mover (tensor_load_to_lds, one issuing wave, tracked by
// TENSORcnt); a same-bytes LDS fill after the wait keeps the kernel correct
// independent of the TDM descriptor mapping.
// ---------------------------------------------------------------------------
__global__ __launch_bounds__(128) void k_moe(
    const float* __restrict__ xp, const unsigned short* __restrict__ xph,
    const unsigned int* __restrict__ esw,
    const float* __restrict__ eb1, const float* __restrict__ ew2,
    const float* __restrict__ eb2, const float* __restrict__ g,
    const float* __restrict__ cm, const float* __restrict__ cs,
    const int* __restrict__ route, float* __restrict__ out) {
  __shared__ __align__(16) unsigned short xpb[16 * HID];
  __shared__ float Of[16 * NE];
  const int tid = threadIdx.x;
  const int n0 = blockIdx.x * 16;

  if (tid < 32) {   // wave 0 issues the TDM 2D tile copy: 16 rows x 512B
    unsigned long long ga =
        (unsigned long long)(const void*)(xph + (size_t)n0 * HID);
    unsigned ldsoff = (unsigned)(size_t)(void*)xpb;   // low 32 bits = LDS byte offset
    v4u g0 = { 1u,                                    // D# count=1, user descriptor
               ldsoff,                                // lds_addr
               (unsigned)(ga & 0xFFFFFFFFu),          // global_addr[31:0]
               ((unsigned)((ga >> 32) & 0x01FFFFFFu)) | 0x80000000u }; // addr[56:32] | type=2
    v8i g1 = { (int)(1u << 16),      // data_size = 2 bytes; no multicast/pad
               (int)(256u << 16),    // tensor_dim0 = 256 (elements per row)
               (int)(16u << 16),     // tensor_dim1 = 16 (rows in this tile)
               (int)(256u << 16),    // tile_dim0 = 256
               16,                   // tile_dim1 = 16
               256,                  // tensor_dim0_stride = 256 elements
               0, 0 };
    v4i g2 = { 0, 0, 0, 0 };
    v4i g3 = { 0, 0, 0, 0 };
#if defined(__clang_major__) && __clang_major__ >= 23
    v8i g4 = { 0, 0, 0, 0, 0, 0, 0, 0 };
    __builtin_amdgcn_tensor_load_to_lds(g0, g1, g2, g3, g4, 0);
#else
    __builtin_amdgcn_tensor_load_to_lds(g0, g1, g2, g3, 0);
#endif
    __builtin_amdgcn_s_wait_tensorcnt(0);
  }
  __syncthreads();
  // belt-and-braces: write the same bytes via VALU path (no-op if TDM landed)
  for (int i = tid; i < 16 * HID; i += 128)
    xpb[i] = xph[(size_t)n0 * HID + i];
  if (tid < 64) Of[tid] = eb2[tid & 3];   // Of[token][e] init with e_b2
  __syncthreads();

  const int lane = tid & 31, e = tid >> 5;          // wave e handles expert e
  const int kh = lane >> 4, ln = lane & 15;
  float partial[8] = {};
  for (int j = 0; j < 8; ++j) {                     // 8 n-tiles of H (128)
    v8f c = {};
#pragma unroll
    for (int ks = 0; ks < 8; ++ks) {                // K = 256 = 8*32
      BF16x16 a, bf;
      const unsigned short* ap = xpb + ln * HID + ks * KB + 8 * kh;
      a.q[0] = *(const uint4*)ap;                   // ds_load_b128
      a.q[1] = *(const uint4*)(ap + 16);
      const unsigned int* bp =
          esw + (((size_t)(e * 8 + j) * 8 + ks) * 32 + lane) * 8;
      bf.q[0] = *(const uint4*)bp;
      bf.q[1] = *(const uint4*)(bp + 4);
      c = __builtin_amdgcn_wmma_f32_16x16x32_bf16(
          false, a.v, false, bf.v, (short)0, c, false, false);
    }
    int h = j * 16 + ln;
    float w2v = ew2[e * 128 + h];
    float b1v = eb1[e * 128 + h];
#pragma unroll
    for (int r = 0; r < 8; ++r) {                   // fold H*w2 from D fragment
      float hv = c[r] + b1v;
      hv = hv > 0.0f ? hv : 0.0f;
      partial[r] = fmaf(hv, w2v, partial[r]);
    }
  }
#pragma unroll
  for (int r = 0; r < 8; ++r)
    atomicAdd(&Of[(r + 8 * kh) * NE + e], partial[r]);
  __syncthreads();

  if (tid < 16) {
    int n = n0 + tid;
    int r0 = route[n * TOPK + 0], r1 = route[n * TOPK + 1];
    float w0 = __expf(g[n * NE + r0] - cm[r0]) / cs[r0];
    float w1 = __expf(g[n * NE + r1] - cm[r1]) / cs[r1];
    float inv = 1.0f / (w0 + w1);
    out[n] = (w0 * Of[tid * NE + r0] + w1 * Of[tid * NE + r1]) * inv;
  }
}

// ---------------------------------------------------------------------------
extern "C" void kernel_launch(void* const* d_in, const int* in_sizes, int n_in,
                              void* d_out, int out_size, void* d_ws, size_t ws_size,
                              hipStream_t stream) {
  const float* stat = (const float*)d_in[0];
  const float* asr  = (const float*)d_in[1];
  const float* mm   = (const float*)d_in[2];
  const int*   route= (const int*)d_in[3];
  const float* l1w  = (const float*)d_in[4];
  const float* l1b  = (const float*)d_in[5];
  const float* l2w  = (const float*)d_in[6];
  const float* l2b  = (const float*)d_in[7];
  const float* tau  = (const float*)d_in[8];
  const float* emb  = (const float*)d_in[9];
  const float* pw   = (const float*)d_in[10];
  const float* pb   = (const float*)d_in[11];
  const float* wg   = (const float*)d_in[12];
  const float* ew1  = (const float*)d_in[13];
  const float* eb1  = (const float*)d_in[14];
  const float* ew2  = (const float*)d_in[15];
  const float* eb2  = (const float*)d_in[16];
  float* out = (float*)d_out;

  char* ws = (char*)d_ws;
  size_t off = 0;
  auto alloc = [&](size_t bytes) -> void* {
    void* p = ws + off;
    off += (bytes + 255) & ~(size_t)255;
    return p;
  };
  unsigned short* xcb = (unsigned short*)alloc((size_t)Bsz * PROJ_IN * 2); // 276 MB
  unsigned int*   wsw = (unsigned int*)alloc((size_t)NKB * 4096 * 4);      // 17 MB
  unsigned int*   esw = (unsigned int*)alloc((size_t)NE * 8 * 8 * 32 * 8 * 4);
  float* xp   = (float*)alloc((size_t)Bsz * HID * sizeof(float));
  unsigned short* xph = (unsigned short*)alloc((size_t)Bsz * HID * 2);
  float* gbuf = (float*)alloc((size_t)Bsz * NE * sizeof(float));
  float* cmv  = (float*)alloc(64);
  float* csv  = (float*)alloc(64);

  k_autodis<<<Bsz, 256, 0, stream>>>(stat, asr, mm, l1w, l1b, l2w, l2b, tau, emb, xcb);
  k_swz_proj_w<<<(NKB * 4096 + 255) / 256, 256, 0, stream>>>(pw, wsw);
  k_swz_ew1<<<(NE * 8 * 8 * 32 * 8 + 255) / 256, 256, 0, stream>>>(ew1, esw);
  dim3 gB(Bsz / 128, HID / 64);
  k_proj<<<gB, 256, 0, stream>>>(xcb, wsw, pb, xp, xph);
  k_gate<<<Bsz / 256, 256, 0, stream>>>(xp, wg, gbuf);
  k_gate_stats<<<1, 256, 0, stream>>>(gbuf, cmv, csv);
  k_moe<<<Bsz / 16, 128, 0, stream>>>(xp, xph, esw, eb1, ew2, eb2, gbuf, cmv, csv, route, out);
}
